// PolymorphicDense_66116726555096
// MI455X (gfx1250) — matrix operand order repaired and checked by reading, compiled, not run on hardware
//
#include <hip/hip_runtime.h>
#include <hip/hip_bf16.h>
#include <cstdint>

typedef _Float16 v16h __attribute__((ext_vector_type(16)));
typedef _Float16 v8h  __attribute__((ext_vector_type(8)));
typedef float    v8f  __attribute__((ext_vector_type(8)));
typedef float    v4f  __attribute__((ext_vector_type(4)));
typedef int      v4i  __attribute__((ext_vector_type(4)));

#define B_ROWS 4096
#define DD     1024
#define UU     1024
#define MODES  8
#define KTOT   (MODES * DD)     // 8192
#define BM     128
#define BN     128
#define BK     64
#define NT     (KTOT / BK)      // 128 k-steps
#define AROW   72               // padded LDS row length in halves (BK + 8); 144B = 9x16B

// ---- async global->LDS (CDNA5) with safe fallback --------------------------
#if defined(__has_builtin)
#  if __has_builtin(__builtin_amdgcn_global_load_async_to_lds_b128)
#    define USE_ASYNC 1
#  endif
#endif
#ifndef USE_ASYNC
#  define USE_ASYNC 0
#endif

__device__ __forceinline__ void async_copy16(const _Float16* g, _Float16* l) {
#if USE_ASYNC
    typedef __attribute__((address_space(1))) v4i* g4p;
    typedef __attribute__((address_space(3))) v4i* l4p;
    __builtin_amdgcn_global_load_async_to_lds_b128(
        (g4p)(uintptr_t)g, (l4p)(uint32_t)(uintptr_t)l, 0, 0);
#else
    *(v8h*)l = *(const v8h*)g;
#endif
}

__device__ __forceinline__ void wait_async0() {
#if USE_ASYNC
#  if __has_builtin(__builtin_amdgcn_s_wait_asynccnt)
    __builtin_amdgcn_s_wait_asynccnt(0);
#  else
    asm volatile("s_wait_asynccnt 0x0" ::: "memory");
#  endif
#endif
}

// ---------------------------------------------------------------------------
// Kernel 1: per-row similarity.
// ---------------------------------------------------------------------------
__global__ __launch_bounds__(256) void sim_kernel(
    const float* __restrict__ x,
    const float* __restrict__ sens_p,
    const float* __restrict__ key_kernel,   // [DD][3]
    const float* __restrict__ key_bias,     // [3]
    const float* __restrict__ keys_map,     // [MODES][3]
    float* __restrict__ sim)                // [B_ROWS][MODES]
{
    const int wave = threadIdx.x >> 5;
    const int lane = threadIdx.x & 31;
    const int row  = blockIdx.x * 8 + wave;

    const float* xr = x + (size_t)row * DD;
    float a0 = 0.f, a1 = 0.f, a2 = 0.f;
    #pragma unroll 4
    for (int i = 0; i < DD / 32; ++i) {
        int   d  = i * 32 + lane;
        float xv = xr[d];
        a0 += xv * key_kernel[d * 3 + 0];
        a1 += xv * key_kernel[d * 3 + 1];
        a2 += xv * key_kernel[d * 3 + 2];
    }
    #pragma unroll
    for (int off = 16; off > 0; off >>= 1) {
        a0 += __shfl_down(a0, off, 32);
        a1 += __shfl_down(a1, off, 32);
        a2 += __shfl_down(a2, off, 32);
    }
    a0 = __shfl(a0, 0, 32);
    a1 = __shfl(a1, 0, 32);
    a2 = __shfl(a2, 0, 32);

    if (lane < MODES) {
        const int   m = lane;
        const float s = sens_p[0];
        float k0 = a0 + key_bias[0] - keys_map[m * 3 + 0];
        float k1 = a1 + key_bias[1] - keys_map[m * 3 + 1];
        float k2 = a2 + key_bias[2] - keys_map[m * 3 + 2];
        float dist = sqrtf(k0 * k0 + k1 * k1 + k2 * k2) * s;
        sim[(size_t)row * MODES + m] = 1.0f / (dist + 1.0f);
    }
}

// ---------------------------------------------------------------------------
// Kernel 2: x (f32) -> x16 (f16), straight copy-convert.
// ---------------------------------------------------------------------------
__global__ __launch_bounds__(256) void cvt_x_kernel(
    const float* __restrict__ x, _Float16* __restrict__ x16)
{
    const size_t i = ((size_t)blockIdx.x * 256 + threadIdx.x) * 8;
    v4f a = *(const v4f*)(x + i);
    v4f b = *(const v4f*)(x + i + 4);
    v8h h = { (_Float16)a.x, (_Float16)a.y, (_Float16)a.z, (_Float16)a.w,
              (_Float16)b.x, (_Float16)b.y, (_Float16)b.z, (_Float16)b.w };
    *(v8h*)(x16 + i) = h;
}

// ---------------------------------------------------------------------------
// Kernel 3: kernels [KTOT][UU] f32  ->  kernT [UU][KTOT] f16  (LDS-tiled 64x64)
// f32 source is dead after this pass -> nontemporal loads keep it out of L2.
// ---------------------------------------------------------------------------
__global__ __launch_bounds__(256) void transpose_kernel(
    const float* __restrict__ kern, _Float16* __restrict__ kernT)
{
    __shared__ __align__(16) _Float16 tile[64][80];   // pad to 160B rows
    const int k0 = blockIdx.x * 64;
    const int u0 = blockIdx.y * 64;
    const int t  = threadIdx.x;

    #pragma unroll
    for (int p = 0; p < 4; ++p) {
        const int r = (t >> 4) + p * 16;          // k row
        const int c = (t & 15) * 4;               // u col
        v4f v = __builtin_nontemporal_load(
            (const v4f*)(kern + (size_t)(k0 + r) * UU + u0 + c));
        tile[c + 0][r] = (_Float16)v.x;
        tile[c + 1][r] = (_Float16)v.y;
        tile[c + 2][r] = (_Float16)v.z;
        tile[c + 3][r] = (_Float16)v.w;
    }
    __syncthreads();

    const int u  = t >> 2;                        // 0..63
    const int s0 = (t & 3) * 16;                  // half offset
    v8h h0 = *(v8h*)&tile[u][s0];
    v8h h1 = *(v8h*)&tile[u][s0 + 8];
    _Float16* dst = kernT + (size_t)(u0 + u) * KTOT + k0 + s0;
    *(v8h*)dst       = h0;
    *(v8h*)(dst + 8) = h1;
}

// ---------------------------------------------------------------------------
// Kernel 4: GEMM  out = (1/8) * [ (sim ⊗ x16) @ kernT^T + sim @ biases ]
// Pure f16 operands; sim scale folded into A tile store (once per element);
// B tiles DMA'd via async global->LDS; double-buffered LDS; BK=64 so each
// barrier covers 16 WMMAs per wave.
// ---------------------------------------------------------------------------
__global__ __launch_bounds__(256) void poly_gemm(
    const _Float16* __restrict__ x16,     // [B_ROWS][DD]
    const _Float16* __restrict__ kernT,   // [UU][KTOT]
    const float*    __restrict__ biases,  // [MODES][UU]
    const float*    __restrict__ sim,     // [B_ROWS][MODES]
    float* __restrict__ out)              // [B_ROWS][UU]
{
    __shared__ __align__(16) _Float16 As[2 * BM * AROW];   // 36 KB
    __shared__ __align__(16) _Float16 Bs[2 * BN * AROW];   // 36 KB
    __shared__ float simS[BM * MODES];                     // 4 KB
    __shared__ float biasS[MODES * BN];                    // 4 KB

    const int tid   = threadIdx.x;
    const int lane  = tid & 31;
    const int l16   = lane & 15;
    const int lh    = lane >> 4;
    const int wave  = tid >> 5;
    const int waveM = wave >> 2;       // 0..1
    const int waveN = wave & 3;        // 0..3
    const int mBase = blockIdx.y * BM;
    const int nBase = blockIdx.x * BN;

    // loader mapping: 2 threads per tile row, 64B (32 halves) each
    const int lrow = tid >> 1;         // 0..127
    const int lq   = (tid & 1) * 32;   // half-offset within 64-wide k slab

    // ---- stage sim rows (128x8) and bias columns (8x128) ----
    {
        const v4f* sp = (const v4f*)(sim + (size_t)mBase * MODES);
        ((v4f*)simS)[tid] = sp[tid];
        const int bm = tid >> 5;
        const int bc = (tid & 31) * 4;
        const v4f* bp = (const v4f*)(biases + (size_t)bm * UU + nBase + bc);
        *((v4f*)(biasS + bm * BN + bc)) = bp[0];
    }

    v8f acc[4][2];
    #pragma unroll
    for (int i = 0; i < 4; ++i)
        #pragma unroll
        for (int j = 0; j < 2; ++j)
            acc[i][j] = (v8f)(0.0f);

    v8h aR[4];

    auto issueB = [&](int buf, int kt) {
        const _Float16* g = kernT + (size_t)(nBase + lrow) * KTOT + kt + lq;
        _Float16* l = Bs + buf * BN * AROW + lrow * AROW + lq;
        async_copy16(g,      l);
        async_copy16(g + 8,  l + 8);
        async_copy16(g + 16, l + 16);
        async_copy16(g + 24, l + 24);
    };
    auto loadA = [&](int kt) {
        const int d0 = kt & (DD - 1);
        const _Float16* g = x16 + (size_t)(mBase + lrow) * DD + d0 + lq;
        #pragma unroll
        for (int q = 0; q < 4; ++q)
            aR[q] = *(const v8h*)(g + q * 8);
    };
    auto scaleStoreA = [&](int buf, int m) {
        const _Float16 sh = (_Float16)simS[lrow * MODES + m];
        _Float16* l = As + buf * BM * AROW + lrow * AROW + lq;
        #pragma unroll
        for (int q = 0; q < 4; ++q) {
            v8h s = aR[q] * sh;
            *(v8h*)(l + q * 8) = s;
        }
    };

    auto compute = [&](int buf) {
        const v8h* pA = (const v8h*)(As + buf * BM * AROW);  // 9 v8h per row
        const v8h* pB = (const v8h*)(Bs + buf * BN * AROW);
        #pragma unroll
        for (int kk = 0; kk < 2; ++kk) {          // two 32-deep sub-steps
            const int ko = kk * 4;                // v8h offset within row
            v16h afr[4];
            #pragma unroll
            for (int i = 0; i < 4; ++i) {
                const int rl = waveM * 64 + i * 16 + l16;
                // lanes 0-15: K=0..7 & 16..23 ; lanes 16-31: K=8..15 & 24..31
                v8h lo = pA[rl * 9 + ko + lh];
                v8h hi = pA[rl * 9 + ko + 2 + lh];
                afr[i] = __builtin_shufflevector(lo, hi,
                          0, 1, 2, 3, 4, 5, 6, 7, 8, 9, 10, 11, 12, 13, 14, 15);
            }
            v16h bfr[2];
            #pragma unroll
            for (int j = 0; j < 2; ++j) {
                const int cl = waveN * 32 + j * 16 + l16;
                // lane holds column n: lanes 0-15 K=0..7 & 16..23 ; lanes 16-31 K=8..15 & 24..31
                v8h lo = pB[cl * 9 + ko + lh];
                v8h hi = pB[cl * 9 + ko + 2 + lh];
                bfr[j] = __builtin_shufflevector(lo, hi,
                          0, 1, 2, 3, 4, 5, 6, 7, 8, 9, 10, 11, 12, 13, 14, 15);
            }
            #pragma unroll
            for (int i = 0; i < 4; ++i)
                #pragma unroll
                for (int j = 0; j < 2; ++j)
                    acc[i][j] = __builtin_amdgcn_wmma_f32_16x16x32_f16(
                        false, afr[i], false, bfr[j], (short)0, acc[i][j],
                        false, false);
        }
    };

    // ---- prologue ----
    issueB(0, 0);
    loadA(0);
    __syncthreads();            // simS/biasS staged
    scaleStoreA(0, 0);
    wait_async0();
    __syncthreads();

    // ---- double-buffered main loop ----
    #pragma unroll 1
    for (int t = 0; t < NT; ++t) {
        const int cur = t & 1;
        const int nxt = cur ^ 1;
        if (t + 1 < NT) {
            issueB(nxt, (t + 1) * BK);   // DMA next B tile
            loadA((t + 1) * BK);         // next A slab into regs
        }
        compute(cur);
        if (t + 1 < NT)
            scaleStoreA(nxt, (t + 1) >> 4);  // mode m = ((t+1)*BK)/1024
        wait_async0();
        __syncthreads();
    }

    // ---- epilogue: + sim@biases, * 1/M ----
    #pragma unroll
    for (int i = 0; i < 4; ++i) {
        #pragma unroll
        for (int j = 0; j < 2; ++j) {
            const int rl0 = waveM * 64 + i * 16 + 8 * lh;
            const int cl  = waveN * 32 + j * 16 + l16;
            const int col = nBase + cl;
            #pragma unroll
            for (int e = 0; e < 8; ++e) {
                const int rl = rl0 + e;
                float bsum = 0.f;
                #pragma unroll
                for (int m = 0; m < MODES; ++m)
                    bsum += simS[rl * MODES + m] * biasS[m * BN + cl];
                out[(size_t)(mBase + rl) * UU + col] =
                    0.125f * (acc[i][j][e] + bsum);
            }
        }
    }
}

// ---------------------------------------------------------------------------
extern "C" void kernel_launch(void* const* d_in, const int* in_sizes, int n_in,
                              void* d_out, int out_size, void* d_ws, size_t ws_size,
                              hipStream_t stream) {
    (void)in_sizes; (void)n_in; (void)out_size; (void)ws_size;
    const float* x          = (const float*)d_in[0];
    const float* sens       = (const float*)d_in[1];
    const float* key_kernel = (const float*)d_in[2];
    const float* key_bias   = (const float*)d_in[3];
    const float* keys_map   = (const float*)d_in[4];
    const float* kernels    = (const float*)d_in[5];
    const float* biases     = (const float*)d_in[6];
    float* out = (float*)d_out;

    // workspace layout
    char* ws = (char*)d_ws;
    float*    sim   = (float*)ws;                              // 128 KB
    _Float16* x16   = (_Float16*)(ws + 131072);                // 8 MB
    _Float16* kernT = (_Float16*)(ws + 131072 + 8388608);      // 16 MB

    sim_kernel<<<B_ROWS / 8, 256, 0, stream>>>(x, sens, key_kernel, key_bias,
                                               keys_map, sim);
    cvt_x_kernel<<<(B_ROWS * DD) / (256 * 8), 256, 0, stream>>>(x, x16);
    {
        dim3 tg(KTOT / 64, UU / 64);   // 128 x 16
        transpose_kernel<<<tg, 256, 0, stream>>>(kernels, kernT);
    }
    {
        dim3 grid(UU / BN, B_ROWS / BM);   // 8 x 32 = 256 workgroups
        poly_gemm<<<grid, 256, 0, stream>>>(x16, kernT, biases, sim, out);
    }
}
